// Attention_16441134809282
// MI455X (gfx1250) — matrix operand order repaired and checked by reading, not compiled, not run on hardware
//
#include <hip/hip_runtime.h>
#include <hip/hip_bf16.h>

// ---------------- problem constants ----------------
#define TT      2048
#define DD      3072
#define NHEAD   16
#define KHEAD   8
#define HDIM    256
#define QCOLS   4096   // N*H
#define KVCOLS  2048   // K*H
#define TOTCOLS 8192   // QCOLS + 2*KVCOLS
#define WIN     1024
#define SOFTCAP 50.0f
#define KMASK  (-3.0e38f)

typedef __bf16 v16bf  __attribute__((ext_vector_type(16)));
typedef __bf16 bf16x8 __attribute__((ext_vector_type(8)));
typedef __bf16 bf16x4 __attribute__((ext_vector_type(4)));
typedef float  v8f    __attribute__((ext_vector_type(8)));

// Assemble a 16-element bf16 WMMA fragment from two contiguous 8-element chunks.
__device__ __forceinline__ v16bf mk16(const __bf16* p0, const __bf16* p1) {
    bf16x8 a = *(const bf16x8*)p0;
    bf16x8 b = *(const bf16x8*)p1;
    v16bf r;
#pragma unroll
    for (int i = 0; i < 8; ++i) { r[i] = a[i]; r[i + 8] = b[i]; }
    return r;
}

__device__ __forceinline__ v8f wmma_bf16(v16bf a, v16bf b, v8f c) {
    return __builtin_amdgcn_wmma_f32_16x16x32_bf16(false, a, false, b,
                                                   (short)0, c, false, false);
}

// cheap tanh: (e-1)/(e+1) with e = exp(2x), arg clamped so e never overflows
__device__ __forceinline__ float fast_tanh(float x) {
    float z = fminf(fmaxf(2.0f * x, -20.0f), 20.0f);
    float e = __expf(z);
    return (e - 1.0f) / (e + 1.0f);
}

// =====================================================================
// Kernel 1: fused QKV projection GEMM.
// C[T, 8192] = X[T, 3072] * Wcat[3072, 8192]   (fp32 in, bf16 WMMA, fp32 out)
// 256 thr / 8 waves, 64x64 C tile, BK=32: 2 WMMAs per wave per K-step.
// Segment (q/k/v) choice is block-uniform -> no divergent epilogue.
// =====================================================================
__global__ void __launch_bounds__(256)
qkv_gemm(const float* __restrict__ x, const float* __restrict__ wq,
         const float* __restrict__ wkv,
         float* __restrict__ qf, float* __restrict__ kf, float* __restrict__ vf)
{
    __shared__ __bf16 lA[64][32];   // [m][k]  64B row stride
    __shared__ __bf16 lBt[64][40];  // [n][k]  80B row stride (16B aligned frags)

    const int bm = blockIdx.y, bn = blockIdx.x;
    const int cb = bn * 64;                    // global column base (within one head)

    // block-uniform segment decode: weight base, output base, output stride
    const float* wbase;
    float* dbase;
    int    dstride, dcol;
    if (cb < QCOLS) {
        wbase = wq + (size_t)(cb >> 8) * DD * HDIM + (cb & 255);
        dbase = qf; dstride = QCOLS; dcol = cb;
    } else if (cb < QCOLS + KVCOLS) {
        wbase = wkv + (size_t)((cb - QCOLS) >> 8) * DD * HDIM + (cb & 255);
        dbase = kf; dstride = KVCOLS; dcol = cb - QCOLS;
    } else {
        wbase = wkv + (size_t)(KHEAD + ((cb - QCOLS - KVCOLS) >> 8)) * DD * HDIM + (cb & 255);
        dbase = vf; dstride = KVCOLS; dcol = cb - QCOLS - KVCOLS;
    }

    const int tid = threadIdx.x;
    const int lane = tid & 31, wave = tid >> 5;
    const int mt = wave & 3, nt = wave >> 2;
    const int lrow = lane & 15, half = lane >> 4;
    const int kb = half * 8, kb16 = half * 16;

    const int ar  = tid >> 3;            // 0..31 (A rows ar, ar+32)
    const int ac4 = (tid & 7) * 4;       // 0..28
    const int bcol = tid & 63;           // B column within tile
    const int bk8  = (tid >> 6) * 8;     // B k-chunk base (0,8,16,24)

    v8f acc0 = {}, acc1 = {};
    for (int k0 = 0; k0 < DD; k0 += 32) {
#pragma unroll
        for (int rr = 0; rr < 64; rr += 32) {
            const float4 av = *(const float4*)(x + (size_t)(bm * 64 + ar + rr) * DD + k0 + ac4);
            bf16x4 cv;
            cv[0] = (__bf16)av.x; cv[1] = (__bf16)av.y;
            cv[2] = (__bf16)av.z; cv[3] = (__bf16)av.w;
            *(bf16x4*)&lA[ar + rr][ac4] = cv;
        }
        {
            bf16x8 cv;
#pragma unroll
            for (int i = 0; i < 8; ++i)
                cv[i] = (__bf16)wbase[(size_t)(k0 + bk8 + i) * HDIM + bcol];
            *(bf16x8*)&lBt[bcol][bk8] = cv;
        }
        __syncthreads();
        const int arow = mt * 16 + lrow;
        v16bf a = mk16(&lA[arow][kb16], &lA[arow][kb16 + 8]);
        const int bc = nt * 32 + lrow;
        v16bf b0 = mk16(&lBt[bc][kb16],      &lBt[bc][kb16 + 8]);
        v16bf b1 = mk16(&lBt[bc + 16][kb16], &lBt[bc + 16][kb16 + 8]);
        acc0 = wmma_bf16(a, b0, acc0);
        acc1 = wmma_bf16(a, b1, acc1);
        __syncthreads();
    }

#pragma unroll
    for (int sub = 0; sub < 2; ++sub) {
        const v8f acc = sub ? acc1 : acc0;
        const int g = dcol + nt * 32 + sub * 16 + lrow;
#pragma unroll
        for (int j = 0; j < 8; ++j) {
            const int m = j + half * 8;
            const int t = bm * 64 + mt * 16 + m;
            dbase[(size_t)t * dstride + g] = acc[j];
        }
    }
}

// =====================================================================
// Kernel 2: RMS-norm (+scale) + RoPE, fp32 -> bf16.
// grid (T, 32): rows 0..15 = q heads, 16..23 = k heads, 24..31 = v heads.
// q/k written row-major [t][head][h]; V written PRE-TRANSPOSED as
// v_t[kk*256+h][t] so flash attention's P*V B-fragments are contiguous.
// =====================================================================
__global__ void __launch_bounds__(128)
norm_rope(const float* __restrict__ qf, const float* __restrict__ kf,
          const float* __restrict__ vf,
          const float* __restrict__ q_scale, const float* __restrict__ k_scale,
          __bf16* __restrict__ q_bf, __bf16* __restrict__ k_bf,
          __bf16* __restrict__ v_t)
{
    const int t = blockIdx.x;
    const int row = blockIdx.y;
    const int h = threadIdx.x;

    const float* src; const float* scale; bool rope;
    if (row < 16) {
        src = qf + (size_t)t * QCOLS + row * HDIM;
        scale = q_scale; rope = true;
    } else if (row < 24) {
        src = kf + (size_t)t * KVCOLS + (row - 16) * HDIM;
        scale = k_scale; rope = true;
    } else {
        src = vf + (size_t)t * KVCOLS + (row - 24) * HDIM;
        scale = nullptr; rope = false;
    }

    const float v1 = src[h], v2 = src[h + 128];
    __shared__ float red[128];
    red[h] = v1 * v1 + v2 * v2;
    __syncthreads();
    for (int s = 64; s > 0; s >>= 1) {
        if (h < s) red[h] += red[h + s];
        __syncthreads();
    }
    const float inv = rsqrtf(red[0] * (1.0f / 256.0f) + 1e-6f);
    float y1 = v1 * inv, y2 = v2 * inv;
    if (scale) { y1 *= (1.0f + scale[h]); y2 *= (1.0f + scale[h + 128]); }
    if (rope) {
        const float ts = powf(10000.0f, (float)h * (1.0f / 128.0f));
        float s_, c_;
        sincosf((float)t / ts, &s_, &c_);
        const float o1 = y1 * c_ - y2 * s_;
        const float o2 = y2 * c_ + y1 * s_;
        y1 = o1; y2 = o2;
    }

    if (row < 16) {
        __bf16* dst = q_bf + (size_t)t * QCOLS + row * HDIM;
        dst[h] = (__bf16)y1; dst[h + 128] = (__bf16)y2;
    } else if (row < 24) {
        __bf16* dst = k_bf + (size_t)t * KVCOLS + (row - 16) * HDIM;
        dst[h] = (__bf16)y1; dst[h + 128] = (__bf16)y2;
    } else {
        const int kk = row - 24;
        v_t[(size_t)(kk * HDIM + h)       * TT + t] = (__bf16)y1;
        v_t[(size_t)(kk * HDIM + h + 128) * TT + t] = (__bf16)y2;
    }
}

// =====================================================================
// Kernel 3: flash attention, one wave per (16-query tile, head).
// Sliding window 1024 + causal + tanh soft-cap, online softmax in fp32.
// QK^T: B-frags contiguous from k_bf.  P*V: B-frags contiguous from v_t.
// Only the P tile round-trips through LDS (C-layout -> A-layout).
// =====================================================================
__global__ void __launch_bounds__(32)
flash_attn(const __bf16* __restrict__ q_bf, const __bf16* __restrict__ k_bf,
           const __bf16* __restrict__ v_t, __bf16* __restrict__ enc_bf)
{
    const int qt = blockIdx.x;
    const int n  = blockIdx.y;
    const int kk = n >> 1;                 // GQA: G = N/K = 2
    const int t0 = qt * 16;
    const int lane = threadIdx.x;
    const int lrow = lane & 15, half = lane >> 4;
    const int kb = half * 8, kb16 = half * 16;

    __shared__ __bf16 lP[16][32];          // P tile for A-fragment transpose

    // Resident Q A-fragments (16 t x 256 h as 8 chunks of K=32)
    v16bf aq[8];
    const __bf16* qrow = q_bf + (size_t)(t0 + lrow) * QCOLS + n * HDIM;
#pragma unroll
    for (int c = 0; c < 8; ++c)
        aq[c] = mk16(qrow + c * 32 + kb16, qrow + c * 32 + kb16 + 8);

    float m_row[8], l_row[8];
#pragma unroll
    for (int j = 0; j < 8; ++j) { m_row[j] = KMASK; l_row[j] = 0.0f; }
    v8f acc[16];
#pragma unroll
    for (int ht = 0; ht < 16; ++ht) { v8f z = {}; acc[ht] = z; }

    int sbeg = t0 - (WIN - 1);
    if (sbeg < 0) sbeg = 0;
    sbeg &= ~31;                           // 32-aligned; chunks never cross T=2048
    const int send = t0 + 15;

    for (int s0 = sbeg; s0 <= send; s0 += 32) {
        // prefetch next K/V chunks into cache
        {
            const int sp = (s0 + 32 + lane < TT) ? (s0 + 32 + lane) : (TT - 1);
            __builtin_prefetch(k_bf + (size_t)sp * KVCOLS + kk * HDIM, 0, 1);
            __builtin_prefetch(v_t + (size_t)(kk * HDIM + lane * 8) * TT + s0 + 32, 0, 1);
        }

        // ---- logits: two 16x16 tiles over the 32-key chunk ----
        v8f lg[2];
#pragma unroll
        for (int sub = 0; sub < 2; ++sub) {
            v8f z = {};
            lg[sub] = z;
            const __bf16* krow = k_bf + (size_t)(s0 + sub * 16 + lrow) * KVCOLS + kk * HDIM;
#pragma unroll
            for (int c = 0; c < 8; ++c) {
                v16bf bk = mk16(krow + c * 32 + kb16, krow + c * 32 + kb16 + 8);
                lg[sub] = wmma_bf16(aq[c], bk, lg[sub]);
            }
        }

        // ---- soft-cap + causal/window mask ----
#pragma unroll
        for (int sub = 0; sub < 2; ++sub) {
#pragma unroll
            for (int j = 0; j < 8; ++j) {
                const float lv = fast_tanh(lg[sub][j] * (1.0f / SOFTCAP)) * SOFTCAP;
                const int scol = s0 + sub * 16 + lrow;
                const int trow = t0 + j + half * 8;
                const bool ok = (scol <= trow) && (scol > trow - WIN);
                lg[sub][j] = ok ? lv : KMASK;
            }
        }

        // ---- online softmax (rows live within 16-lane halves) ----
        float newm[8], scl[8];
#pragma unroll
        for (int j = 0; j < 8; ++j) {
            float mx = fmaxf(lg[0][j], lg[1][j]);
#pragma unroll
            for (int off = 1; off < 16; off <<= 1)
                mx = fmaxf(mx, __shfl_xor(mx, off));
            newm[j] = fmaxf(m_row[j], mx);
            scl[j] = (newm[j] > -1e37f) ? __expf(m_row[j] - newm[j]) : 1.0f;
            m_row[j] = newm[j];
        }
#pragma unroll
        for (int j = 0; j < 8; ++j) {
            const float p0 = (lg[0][j] > -1e37f) ? __expf(lg[0][j] - newm[j]) : 0.0f;
            const float p1 = (lg[1][j] > -1e37f) ? __expf(lg[1][j] - newm[j]) : 0.0f;
            float rs = p0 + p1;
#pragma unroll
            for (int off = 1; off < 16; off <<= 1)
                rs += __shfl_xor(rs, off);
            l_row[j] = l_row[j] * scl[j] + rs;
            const int m = j + half * 8;
            lP[m][lrow] = (__bf16)p0;
            lP[m][16 + lrow] = (__bf16)p1;
        }
        __syncthreads();

        // ---- rescale accumulator ----
#pragma unroll
        for (int ht = 0; ht < 16; ++ht)
#pragma unroll
            for (int j = 0; j < 8; ++j) acc[ht][j] *= scl[j];

        // ---- P * V (B-frags contiguous from pre-transposed V) ----
        v16bf ap = mk16(&lP[lrow][kb16], &lP[lrow][kb16 + 8]);
#pragma unroll
        for (int ht = 0; ht < 16; ++ht) {
            const __bf16* vrow = v_t + (size_t)(kk * HDIM + ht * 16 + lrow) * TT + s0;
            v16bf bv = mk16(vrow + kb16, vrow + kb16 + 8);
            acc[ht] = wmma_bf16(ap, bv, acc[ht]);
        }
        __syncthreads();
    }

    // ---- normalize and emit enc (bf16, [t][n*256+h]) ----
#pragma unroll
    for (int ht = 0; ht < 16; ++ht) {
#pragma unroll
        for (int j = 0; j < 8; ++j) {
            const int m = j + half * 8;
            const float o = acc[ht][j] / l_row[j];
            enc_bf[(size_t)(t0 + m) * QCOLS + n * HDIM + ht * 16 + lrow] = (__bf16)o;
        }
    }
}

// =====================================================================
// Kernel 4: output projection.
// out[T, 3072] = enc_bf[T, 4096] * w_out[4096, 3072]  (fp32 accumulate)
// 256 thr / 8 waves, 64x64 tile, BK=32.
// =====================================================================
__global__ void __launch_bounds__(256)
out_gemm(const __bf16* __restrict__ enc_bf, const float* __restrict__ w_out,
         float* __restrict__ out)
{
    __shared__ __bf16 lA[64][32];
    __shared__ __bf16 lBt[64][40];

    const int bm = blockIdx.y, bn = blockIdx.x;
    const int tid = threadIdx.x;
    const int lane = tid & 31, wave = tid >> 5;
    const int mt = wave & 3, nt = wave >> 2;
    const int lrow = lane & 15, half = lane >> 4;
    const int kb = half * 8, kb16 = half * 16;

    const int ar  = tid >> 2, ac8 = (tid & 3) * 8;  // A: 64 rows x 4 chunks of 8
    const int bcol = tid & 63, bk8 = (tid >> 6) * 8;

    v8f acc0 = {}, acc1 = {};
    for (int k0 = 0; k0 < QCOLS; k0 += 32) {
        *(bf16x8*)&lA[ar][ac8] =
            *(const bf16x8*)(enc_bf + (size_t)(bm * 64 + ar) * QCOLS + k0 + ac8);
        {
            bf16x8 cv;
#pragma unroll
            for (int i = 0; i < 8; ++i)
                cv[i] = (__bf16)w_out[(size_t)(k0 + bk8 + i) * DD + bn * 64 + bcol];
            *(bf16x8*)&lBt[bcol][bk8] = cv;
        }
        __syncthreads();
        const int arow = mt * 16 + lrow;
        v16bf a = mk16(&lA[arow][kb16], &lA[arow][kb16 + 8]);
        const int bc = nt * 32 + lrow;
        v16bf b0 = mk16(&lBt[bc][kb16],      &lBt[bc][kb16 + 8]);
        v16bf b1 = mk16(&lBt[bc + 16][kb16], &lBt[bc + 16][kb16 + 8]);
        acc0 = wmma_bf16(a, b0, acc0);
        acc1 = wmma_bf16(a, b1, acc1);
        __syncthreads();
    }

#pragma unroll
    for (int sub = 0; sub < 2; ++sub) {
        const v8f acc = sub ? acc1 : acc0;
#pragma unroll
        for (int j = 0; j < 8; ++j) {
            const int m = j + half * 8;
            out[(size_t)(bm * 64 + mt * 16 + m) * DD + bn * 64 + nt * 32 + sub * 16 + lrow] = acc[j];
        }
    }
}

// =====================================================================
// launcher
// =====================================================================
extern "C" void kernel_launch(void* const* d_in, const int* in_sizes, int n_in,
                              void* d_out, int out_size, void* d_ws, size_t ws_size,
                              hipStream_t stream) {
    (void)in_sizes; (void)n_in; (void)out_size; (void)ws_size;
    const float* x       = (const float*)d_in[0];
    // d_in[1]=segment_pos (arange, recomputed), d_in[2]=attn_mask (causal, recomputed)
    const float* w_q     = (const float*)d_in[3];
    const float* w_kv    = (const float*)d_in[4];
    const float* w_out   = (const float*)d_in[5];
    const float* q_scale = (const float*)d_in[6];
    const float* k_scale = (const float*)d_in[7];

    char* ws = (char*)d_ws;
    float*  qf    = (float*)(ws);                       // T*4096 f32 = 32 MB
    float*  kf    = (float*)(ws + 33554432);            // T*2048 f32 = 16 MB
    float*  vf    = (float*)(ws + 50331648);            // T*2048 f32 = 16 MB
    __bf16* q_bf  = (__bf16*)(ws + 67108864);           // T*4096 bf16
    __bf16* k_bf  = (__bf16*)(ws + 83886080);           // T*2048 bf16
    __bf16* v_t   = (__bf16*)(ws + 92274688);           // [K*H][T] bf16 (transposed)
    __bf16* enc   = (__bf16*)(ws + 100663296);          // T*4096 bf16
    float*  out   = (float*)d_out;

    qkv_gemm <<<dim3(TOTCOLS / 64, TT / 64), 256, 0, stream>>>(x, w_q, w_kv, qf, kf, vf);
    norm_rope<<<dim3(TT, 32),               128, 0, stream>>>(qf, kf, vf, q_scale, k_scale,
                                                              q_bf, k_bf, v_t);
    flash_attn<<<dim3(TT / 16, NHEAD),       32, 0, stream>>>(q_bf, k_bf, v_t, enc);
    out_gemm <<<dim3(DD / 64, TT / 64),     256, 0, stream>>>(enc, w_out, out);
}
